// GRU_16552803958911
// MI455X (gfx1250) — matrix operand order repaired and checked by reading, not compiled, not run on hardware
//
#include <hip/hip_runtime.h>
#include <hip/hip_bf16.h>
#include <cmath>

// ---------------------------------------------------------------------------
// GRU (B=64, T=1024, I=64, H=512, L=2) + FC for MI455X (gfx1250, wave32).
// All GEMMs use v_wmma_f32_16x16x32_bf16 (f32 accumulate).
//  - Input-side gate GEMMs are batched over B*T (parallel, compute-bound).
//  - The recurrent chain is 1024 fused step-kernels per layer (GEMM+gates),
//    weights stay hot in the 192MB L2 (1.5MB bf16 per layer).
// Workspace use: ~550 MB (xg buffer in f32 is shared between both layers).
// ---------------------------------------------------------------------------

typedef __attribute__((ext_vector_type(16))) __bf16 bf16x16;
typedef __attribute__((ext_vector_type(8)))  __bf16 bf16x8;
typedef __attribute__((ext_vector_type(8)))  float  f32x8;

#define GB 64      // batch
#define GT 1024    // time steps
#define GI 64      // input size
#define GH 512     // hidden size
#define GG 1536    // 3*H
#define GM (GB*GT) // 65536 rows for batched GEMMs

// ---------------------------------------------------------------------------
// fp32 -> bf16 conversion
__global__ void __launch_bounds__(256) cvt_bf16_kernel(const float* __restrict__ src,
                                                       __bf16* __restrict__ dst, int n) {
    int i = blockIdx.x * 256 + threadIdx.x;
    if (i < n) dst[i] = (__bf16)src[i];
}

__global__ void __launch_bounds__(256) zero_bf16_kernel(__bf16* __restrict__ dst, int n) {
    int i = blockIdx.x * 256 + threadIdx.x;
    if (i < n) dst[i] = (__bf16)0.0f;
}

// ---------------------------------------------------------------------------
// Generic WMMA GEMM:  C[M,N] = A[M,K] * W[N,K]^T + bias[N]   (A,W bf16; C f32)
// Block = 256 threads = 8 waves; block tile = 128(M) x 64(N); wave tile 16x64.
// grid = (M/128, N/64). K multiple of 32.
__global__ void __launch_bounds__(256)
gemm_bf16_wmma(const __bf16* __restrict__ A, const __bf16* __restrict__ W,
               const float* __restrict__ bias, float* __restrict__ C,
               int K, int ldc) {
    const int lane = threadIdx.x & 31;
    const int wave = threadIdx.x >> 5;
    const int half = lane >> 4;     // 0: lanes 0-15, 1: lanes 16-31
    const int l16  = lane & 15;
    const int m0   = blockIdx.x * 128 + wave * 16;
    const int n0   = blockIdx.y * 64;

    const __bf16* Arow = A + (size_t)(m0 + l16) * K + half * 16;

    f32x8 acc[4];
#pragma unroll
    for (int j = 0; j < 4; ++j) {
        float bv = bias[n0 + j * 16 + l16];
#pragma unroll
        for (int v = 0; v < 8; ++v) acc[j][v] = bv;
    }

    for (int k0 = 0; k0 < K; k0 += 32) {
        union { bf16x16 v; bf16x8 h[2]; } a;
        a.h[0] = *(const bf16x8*)(Arow + k0);        // K = k0 + 16*half + [0..7]
        a.h[1] = *(const bf16x8*)(Arow + k0 + 8);    // K = k0 + 16*half + 8 + [0..7]
#pragma unroll
        for (int j = 0; j < 4; ++j) {
            const __bf16* Wp = W + (size_t)(n0 + j * 16 + l16) * K + k0 + half * 16;
            bf16x16 bfrag = *(const bf16x16*)Wp;     // 16 contiguous K values
            acc[j] = __builtin_amdgcn_wmma_f32_16x16x32_bf16(
                false, a.v, false, bfrag, (short)0, acc[j], false, false);
        }
    }

    const int crow = m0 + half * 8;
#pragma unroll
    for (int j = 0; j < 4; ++j) {
        float* Cp = C + (size_t)crow * ldc + n0 + j * 16 + l16;
#pragma unroll
        for (int v = 0; v < 8; ++v) Cp[(size_t)v * ldc] = acc[j][v];
    }
}

// ---------------------------------------------------------------------------
// Fused GRU step:  hg = h_in @ Whh^T + bhh ; gates with xg[:,t,:] ; write h_out, ys[:,t,:]
// grid = 16 blocks; block cb owns hidden columns [cb*32, cb*32+32) for all 3 gates.
// 8 waves: wave tile = 16(M) x 16(N), 3 tiles per wave (24 tiles = 4 mi x 3 gate x 2 nj).
__global__ void __launch_bounds__(256)
gru_step_kernel(const __bf16* __restrict__ hin,   // [64, 512] bf16
                __bf16* __restrict__ hout,        // [64, 512] bf16
                const __bf16* __restrict__ Whh,   // [1536, 512] bf16
                const float* __restrict__ bhh,    // [1536]
                const float* __restrict__ xg,     // [64, T, 1536] f32
                __bf16* __restrict__ ys,          // [64, T, 512] bf16
                int t) {
    __shared__ float hg[3][64][32];

    const int lane = threadIdx.x & 31;
    const int wave = threadIdx.x >> 5;
    const int half = lane >> 4;
    const int l16  = lane & 15;
    const int cb   = blockIdx.x;          // 0..15
    const int m0   = (wave & 3) * 16;

    int gate[3], nj[3];
    f32x8 acc[3];
#pragma unroll
    for (int i = 0; i < 3; ++i) {
        int q = (wave >> 2) + 2 * i;      // 0..5
        gate[i] = q % 3;
        nj[i]   = q / 3;                  // 0..1
        int gn = gate[i] * GH + cb * 32 + nj[i] * 16 + l16;
        float bv = bhh[gn];
#pragma unroll
        for (int v = 0; v < 8; ++v) acc[i][v] = bv;
    }

    const __bf16* Arow = hin + (size_t)(m0 + l16) * GH + half * 16;
    for (int k0 = 0; k0 < GH; k0 += 32) {
        union { bf16x16 v; bf16x8 h[2]; } a;
        a.h[0] = *(const bf16x8*)(Arow + k0);
        a.h[1] = *(const bf16x8*)(Arow + k0 + 8);
#pragma unroll
        for (int i = 0; i < 3; ++i) {
            int gn = gate[i] * GH + cb * 32 + nj[i] * 16 + l16;
            bf16x16 bfrag = *(const bf16x16*)(Whh + (size_t)gn * GH + k0 + half * 16);
            acc[i] = __builtin_amdgcn_wmma_f32_16x16x32_bf16(
                false, a.v, false, bfrag, (short)0, acc[i], false, false);
        }
    }

    // spill hg tiles to LDS
#pragma unroll
    for (int i = 0; i < 3; ++i) {
        int nl  = nj[i] * 16 + l16;
        int row = m0 + half * 8;
#pragma unroll
        for (int v = 0; v < 8; ++v) hg[gate[i]][row + v][nl] = acc[i][v];
    }
    __syncthreads();

    // gate nonlinearities: 64 rows x 32 cols, 8 elements per thread
    const int b  = threadIdx.x >> 2;        // 0..63
    const int c0 = (threadIdx.x & 3) * 8;   // 0,8,16,24
    const float* xgp = xg + ((size_t)b * GT + t) * GG;
#pragma unroll
    for (int e = 0; e < 8; ++e) {
        int jl = c0 + e;
        int j  = cb * 32 + jl;
        float xr = xgp[j];
        float xz = xgp[GH + j];
        float xn = xgp[2 * GH + j];
        float r = 1.0f / (1.0f + expf(-(xr + hg[0][b][jl])));
        float z = 1.0f / (1.0f + expf(-(xz + hg[1][b][jl])));
        float n = tanhf(xn + r * hg[2][b][jl]);
        float hp = (float)hin[(size_t)b * GH + j];
        float hv = (1.0f - z) * n + z * hp;
        __bf16 hb = (__bf16)hv;
        hout[(size_t)b * GH + j] = hb;
        ys[((size_t)b * GT + t) * GH + j] = hb;
    }
}

// ---------------------------------------------------------------------------
extern "C" void kernel_launch(void* const* d_in, const int* in_sizes, int n_in,
                              void* d_out, int out_size, void* d_ws, size_t ws_size,
                              hipStream_t stream) {
    const float* x    = (const float*)d_in[0];
    const float* wih0 = (const float*)d_in[1];
    const float* whh0 = (const float*)d_in[2];
    const float* bih0 = (const float*)d_in[3];
    const float* bhh0 = (const float*)d_in[4];
    const float* wih1 = (const float*)d_in[5];
    const float* whh1 = (const float*)d_in[6];
    const float* bih1 = (const float*)d_in[7];
    const float* bhh1 = (const float*)d_in[8];
    const float* fcw  = (const float*)d_in[9];
    const float* fcb  = (const float*)d_in[10];
    float* out = (float*)d_out;

    // workspace carve-up (256B aligned)
    char* ws = (char*)d_ws;
    size_t off = 0;
    auto alloc = [&](size_t bytes) -> void* {
        void* p = ws + off;
        off = (off + bytes + 255) & ~(size_t)255;
        return p;
    };
    __bf16* x_bf    = (__bf16*)alloc((size_t)GM * GI * 2);
    __bf16* wih0_bf = (__bf16*)alloc((size_t)GG * GI * 2);
    __bf16* whh0_bf = (__bf16*)alloc((size_t)GG * GH * 2);
    __bf16* wih1_bf = (__bf16*)alloc((size_t)GG * GH * 2);
    __bf16* whh1_bf = (__bf16*)alloc((size_t)GG * GH * 2);
    __bf16* fcw_bf  = (__bf16*)alloc((size_t)GI * GH * 2);
    __bf16* hbuf    = (__bf16*)alloc((size_t)2 * GB * GH * 2);
    __bf16* ys0     = (__bf16*)alloc((size_t)GM * GH * 2);
    __bf16* ys1     = (__bf16*)alloc((size_t)GM * GH * 2);
    float*  xg      = (float*)alloc((size_t)GM * GG * 4);  // shared by both layers

    auto cvt = [&](const float* s, __bf16* d, int n) {
        cvt_bf16_kernel<<<(n + 255) / 256, 256, 0, stream>>>(s, d, n);
    };
    cvt(x,    x_bf,    GM * GI);
    cvt(wih0, wih0_bf, GG * GI);
    cvt(whh0, whh0_bf, GG * GH);
    cvt(wih1, wih1_bf, GG * GH);
    cvt(whh1, whh1_bf, GG * GH);
    cvt(fcw,  fcw_bf,  GI * GH);

    // ---- layer 0 ----
    // xg = x @ wih0^T + bih0   (M=65536, N=1536, K=64)
    gemm_bf16_wmma<<<dim3(GM / 128, GG / 64), 256, 0, stream>>>(
        x_bf, wih0_bf, bih0, xg, GI, GG);
    zero_bf16_kernel<<<(2 * GB * GH + 255) / 256, 256, 0, stream>>>(hbuf, 2 * GB * GH);
    for (int t = 0; t < GT; ++t) {
        gru_step_kernel<<<16, 256, 0, stream>>>(
            hbuf + (size_t)(t & 1) * GB * GH,
            hbuf + (size_t)((t + 1) & 1) * GB * GH,
            whh0_bf, bhh0, xg, ys0, t);
    }

    // ---- layer 1 ----
    // xg = ys0 @ wih1^T + bih1   (M=65536, N=1536, K=512)
    gemm_bf16_wmma<<<dim3(GM / 128, GG / 64), 256, 0, stream>>>(
        ys0, wih1_bf, bih1, xg, GH, GG);
    zero_bf16_kernel<<<(2 * GB * GH + 255) / 256, 256, 0, stream>>>(hbuf, 2 * GB * GH);
    for (int t = 0; t < GT; ++t) {
        gru_step_kernel<<<16, 256, 0, stream>>>(
            hbuf + (size_t)(t & 1) * GB * GH,
            hbuf + (size_t)((t + 1) & 1) * GB * GH,
            whh1_bf, bhh1, xg, ys1, t);
    }

    // ---- FC ----
    // out = ys1 @ fcw^T + fcb   (M=65536, N=64, K=512)
    gemm_bf16_wmma<<<dim3(GM / 128, GI / 64), 256, 0, stream>>>(
        ys1, fcw_bf, fcb, out, GH, GI);
}
